// MultiHeadAttentionBlock_12704513261578
// MI455X (gfx1250) — compile-verified, matrix operand order checked by reading, not run on hardware
//
#include <hip/hip_runtime.h>

// ---------------------------------------------------------------------------
// MHA block for MI455X (gfx1250): bf16 WMMA GEMMs + flash-attention.
// D_MODEL=1024, H=16, D_K=64, B=2, S=2048.  All matmuls via
// v_wmma_f32_16x16x32_bf16 (wave32, 16x16 tiles, K=32 per instruction).
// Pure bf16 tile staging uses GLOBAL_LOAD_ASYNC_TO_LDS_B128 (ASYNCcnt path).
// ---------------------------------------------------------------------------

typedef __attribute__((ext_vector_type(16))) __bf16 v16bf;
typedef __attribute__((ext_vector_type(8)))  __bf16 bf16x8;
typedef __attribute__((ext_vector_type(8)))  float  v8f;
typedef __attribute__((ext_vector_type(4)))  int    v4i;

#define DEV __device__ __forceinline__
#define AS1 __attribute__((address_space(1)))
#define AS3 __attribute__((address_space(3)))

#if defined(__has_builtin)
#if __has_builtin(__builtin_amdgcn_global_load_async_to_lds_b128)
#define HAVE_ASYNC_LDS 1
#endif
#endif
#ifndef HAVE_ASYNC_LDS
#define HAVE_ASYNC_LDS 0
#endif

constexpr int D_MODEL = 1024;
constexpr int NH      = 16;
constexpr int DK      = 64;
constexpr int BATCH   = 2;
constexpr int SEQ     = 2048;
constexpr int MTOT    = BATCH * SEQ;   // 4096 rows

// ---------------------------------------------------------------------------
// helpers
// ---------------------------------------------------------------------------
DEV v16bf make_frag(bf16x8 lo, bf16x8 hi) {
  return __builtin_shufflevector(lo, hi, 0, 1, 2, 3, 4, 5, 6, 7,
                                          8, 9, 10, 11, 12, 13, 14, 15);
}

// Load a 16x32 bf16 operand fragment from a row-major [16 x ldm] region.
// Lane l (l<16): rows m=l; lanes 16..31 carry the second K half-chunks,
// matching the CDNA5 16-bit A-operand striping (B operands use the same
// pattern when the matrix is supplied N-major/K-contiguous).
DEV v16bf load_frag(const __bf16* base, int ldm, int kOff) {
  const int lane = threadIdx.x & 31;
  const int m    = lane & 15;
  const int sel  = lane >> 4;
  const __bf16* p = base + m * ldm + kOff + sel * 8;
  bf16x8 lo = *(const bf16x8*)p;          // K = sel*8 .. sel*8+7
  bf16x8 hi = *(const bf16x8*)(p + 16);   // K = 16+sel*8 ..
  return make_frag(lo, hi);
}

DEV v8f wmma_bf16(v16bf a, v16bf b, v8f c) {
  // (neg_a, A, neg_b, B, c_mod, C, reuse_a, reuse_b)
  return __builtin_amdgcn_wmma_f32_16x16x32_bf16(false, a, false, b,
                                                 (short)0, c, false, false);
}

DEV float half16_max(float v) {   // reduce across the 16 lanes of a half-wave
  v = fmaxf(v, __shfl_xor(v, 8, 32));
  v = fmaxf(v, __shfl_xor(v, 4, 32));
  v = fmaxf(v, __shfl_xor(v, 2, 32));
  v = fmaxf(v, __shfl_xor(v, 1, 32));
  return v;
}
DEV float half16_sum(float v) {
  v += __shfl_xor(v, 8, 32);
  v += __shfl_xor(v, 4, 32);
  v += __shfl_xor(v, 2, 32);
  v += __shfl_xor(v, 1, 32);
  return v;
}

// Load 8 elements from global, converting f32 -> bf16 on the fly if needed.
template <bool F32>
DEV bf16x8 ld8cvt(const void* p) {
  if constexpr (F32) {
    const float4* q = (const float4*)p;
    float4 a = q[0], b = q[1];
    bf16x8 r;
    r[0] = (__bf16)a.x; r[1] = (__bf16)a.y; r[2] = (__bf16)a.z; r[3] = (__bf16)a.w;
    r[4] = (__bf16)b.x; r[5] = (__bf16)b.y; r[6] = (__bf16)b.z; r[7] = (__bf16)b.w;
    return r;
  } else {
    return *(const bf16x8*)p;
  }
}

// 16-byte global -> LDS copy: async (ASYNCcnt) path when the toolchain has
// the gfx1250 builtin, else a regular load/ds_store pair.
DEV void cp16_g2l(__bf16* ldst, const __bf16* gsrc) {
#if HAVE_ASYNC_LDS
  v4i* g = (v4i*)const_cast<__bf16*>(gsrc);
  v4i* l = (v4i*)ldst;
  __builtin_amdgcn_global_load_async_to_lds_b128((AS1 v4i*)g, (AS3 v4i*)l,
                                                 0, 0);
#else
  *(bf16x8*)ldst = *(const bf16x8*)gsrc;
#endif
}

DEV void async_wait0() {
#if HAVE_ASYNC_LDS
#if __has_builtin(__builtin_amdgcn_s_wait_asynccnt)
  __builtin_amdgcn_s_wait_asynccnt(0);
#else
  asm volatile("s_wait_asynccnt 0x0" ::: "memory");
#endif
#endif
}

// ---------------------------------------------------------------------------
// GEMM: C[M=4096, N=1024] = A[M,K=1024] @ W[N,K]^T + bias
//   OUT_MODE 0: write bf16 in head layout [B, H, S, DK]   (projections)
//   OUT_MODE 1: write f32 flat [M, N]                     (final output)
// Block = 256 threads (8 waves), tile 128x128, BK = 32.
// Each wave -> 2x4 WMMA tiles (32x64 output).
// ---------------------------------------------------------------------------
template <typename TA, int OUT_MODE>
__global__ __launch_bounds__(256)
void mha_gemm_wmma(const TA* __restrict__ A, const float* __restrict__ W,
                   const float* __restrict__ bias, void* __restrict__ outp) {
  constexpr bool A_F32 = (sizeof(TA) == 4);
  __shared__ __align__(16) __bf16 sA[128 * 32];
  __shared__ __align__(16) __bf16 sB[128 * 32];

  const int tid  = threadIdx.x;
  const int wid  = tid >> 5;
  const int lane = tid & 31;
  const int nl   = lane & 15;
  const int sel  = lane >> 4;
  const int m0   = blockIdx.x * 128;
  const int n0   = blockIdx.y * 128;
  const int wm   = (wid >> 1) * 32;   // wave row offset in tile
  const int wn   = (wid & 1) * 64;    // wave col offset in tile

  v8f acc[2][4];
#pragma unroll
  for (int i = 0; i < 2; ++i)
#pragma unroll
    for (int j = 0; j < 4; ++j)
      acc[i][j] = v8f{0.f, 0.f, 0.f, 0.f, 0.f, 0.f, 0.f, 0.f};

  const int row  = tid >> 1;          // 0..127
  const int part = (tid & 1) * 16;    // 0 or 16 (elements)

  for (int k0 = 0; k0 < D_MODEL; k0 += 32) {
    __syncthreads();  // previous iteration's fragment reads done
    {
      const TA*    ga = A + (size_t)(m0 + row) * D_MODEL + k0 + part;
      const float* gw = W + (size_t)(n0 + row) * D_MODEL + k0 + part;
      if constexpr (A_F32) {
        *(bf16x8*)&sA[row * 32 + part]     = ld8cvt<true>(ga);
        *(bf16x8*)&sA[row * 32 + part + 8] = ld8cvt<true>(ga + 8);
      } else {
        cp16_g2l(&sA[row * 32 + part],     (const __bf16*)ga);
        cp16_g2l(&sA[row * 32 + part + 8], (const __bf16*)ga + 8);
      }
      *(bf16x8*)&sB[row * 32 + part]     = ld8cvt<true>(gw);
      *(bf16x8*)&sB[row * 32 + part + 8] = ld8cvt<true>(gw + 8);
      if (k0 + 32 < D_MODEL) {   // global_prefetch the next K slab
        __builtin_prefetch(ga + 32, 0, 0);
        __builtin_prefetch(gw + 32, 0, 0);
      }
    }
    if constexpr (!A_F32) async_wait0();
    __syncthreads();

    v16bf af[2], bfr[4];
    af[0] = load_frag(&sA[(wm +  0) * 32], 32, 0);
    af[1] = load_frag(&sA[(wm + 16) * 32], 32, 0);
#pragma unroll
    for (int j = 0; j < 4; ++j)
      bfr[j] = load_frag(&sB[(wn + j * 16) * 32], 32, 0);

#pragma unroll
    for (int i = 0; i < 2; ++i)
#pragma unroll
      for (int j = 0; j < 4; ++j)
        acc[i][j] = wmma_bf16(af[i], bfr[j], acc[i][j]);
  }

  // Epilogue: +bias, scatter per the WMMA C layout
  // (VGPR r -> M = r + 8*sel ; lane nl -> N = nl).
#pragma unroll
  for (int j = 0; j < 4; ++j) {
    const int col = n0 + wn + j * 16 + nl;
    const float bv = bias[col];
#pragma unroll
    for (int i = 0; i < 2; ++i) {
#pragma unroll
      for (int r = 0; r < 8; ++r) {
        const int rowg = m0 + wm + i * 16 + r + sel * 8;
        const float v = acc[i][j][r] + bv;
        if (OUT_MODE == 0) {
          const int b = rowg >> 11, s = rowg & (SEQ - 1);
          const int h = col >> 6,  d = col & (DK - 1);
          ((__bf16*)outp)[(((size_t)(b * NH + h) * SEQ) + s) * DK + d] = (__bf16)v;
        } else {
          ((float*)outp)[(size_t)rowg * D_MODEL + col] = v;
        }
      }
    }
  }
}

// ---------------------------------------------------------------------------
// Flash attention: one block = 128 query rows of one (b,h); 8 waves x 16 rows.
// K-block = 64 keys per iteration; online softmax; causal mask analytic.
// qh/kh/vh: bf16 [B*H, S, DK].  out: bf16 [B*S, D_MODEL] (heads merged).
// ---------------------------------------------------------------------------
__global__ __launch_bounds__(256)
void mha_flash_attn_wmma(const __bf16* __restrict__ qh,
                         const __bf16* __restrict__ kh,
                         const __bf16* __restrict__ vh,
                         __bf16* __restrict__ out) {
  __shared__ __align__(16) __bf16 sK[64 * 64];        // [key][d]
  __shared__ __align__(16) __bf16 sVt[64 * 64];       // [d][key] (transposed)
  __shared__ __align__(16) __bf16 sP[8 * 16 * 64];    // per-wave P tiles

  const int tid  = threadIdx.x;
  const int wid  = tid >> 5;
  const int lane = tid & 31;
  const int nl   = lane & 15;
  const int sel  = lane >> 4;

  const int qb = blockIdx.x * 128;
  const int bh = blockIdx.y;            // b*NH + h
  const int q0 = qb + wid * 16;         // this wave's first query row

  const __bf16* Qb = qh + (size_t)bh * SEQ * DK;
  const __bf16* Kb = kh + (size_t)bh * SEQ * DK;
  const __bf16* Vb = vh + (size_t)bh * SEQ * DK;

  // Q fragments for this wave's 16 rows (d = 0..31 and 32..63), kept live.
  v16bf qf[2];
  qf[0] = load_frag(Qb + (size_t)q0 * DK, DK, 0);
  qf[1] = load_frag(Qb + (size_t)q0 * DK, DK, 32);

  v8f o[4];
#pragma unroll
  for (int dt = 0; dt < 4; ++dt)
    o[dt] = v8f{0.f, 0.f, 0.f, 0.f, 0.f, 0.f, 0.f, 0.f};
  float mst[8], lst[8];
#pragma unroll
  for (int r = 0; r < 8; ++r) { mst[r] = -1e30f; lst[r] = 0.f; }

  const int nkb = (qb >> 6) + 2;        // causal: key blocks needed
  for (int kb = 0; kb < nkb; ++kb) {
    __syncthreads();
    // Cooperative load: K tile async row-major, V tile transposed into sVt.
#pragma unroll
    for (int s2 = 0; s2 < 2; ++s2) {
      const int seg  = tid + s2 * 256;      // 0..511
      const int krow = seg >> 3;            // 0..63
      const int kprt = (seg & 7) * 8;       // 0..56
      cp16_g2l(&sK[krow * 64 + kprt],
               Kb + (size_t)(kb * 64 + krow) * DK + kprt);
      bf16x8 vv = *(const bf16x8*)(Vb + (size_t)(kb * 64 + krow) * DK + kprt);
#pragma unroll
      for (int j = 0; j < 8; ++j) sVt[(kprt + j) * 64 + krow] = vv[j];
    }
    async_wait0();
    __syncthreads();

    // scores = Q @ K^T  (4 key tiles of 16, accumulated over d in 2 steps)
    v8f sc[4];
#pragma unroll
    for (int kk = 0; kk < 4; ++kk) {
      sc[kk] = v8f{0.f, 0.f, 0.f, 0.f, 0.f, 0.f, 0.f, 0.f};
      v16bf kf0 = load_frag(&sK[kk * 16 * 64], 64, 0);
      v16bf kf1 = load_frag(&sK[kk * 16 * 64], 64, 32);
      sc[kk] = wmma_bf16(qf[0], kf0, sc[kk]);
      sc[kk] = wmma_bf16(qf[1], kf1, sc[kk]);
    }

    // scale + causal mask
#pragma unroll
    for (int kk = 0; kk < 4; ++kk) {
#pragma unroll
      for (int r = 0; r < 8; ++r) {
        const int rowg = q0 + r + sel * 8;
        const int colg = kb * 64 + kk * 16 + nl;
        const float s = sc[kk][r] * 0.125f;   // 1/sqrt(64)
        sc[kk][r] = (colg <= rowg) ? s : -1e30f;
      }
    }

    // online softmax (row lives across the 16 lanes of a half-wave)
#pragma unroll
    for (int r = 0; r < 8; ++r) {
      float mx = fmaxf(fmaxf(sc[0][r], sc[1][r]), fmaxf(sc[2][r], sc[3][r]));
      mx = half16_max(mx);
      const float mnew  = fmaxf(mst[r], mx);
      const float alpha = __expf(mst[r] - mnew);
      float rs = 0.f;
#pragma unroll
      for (int kk = 0; kk < 4; ++kk) {
        const float p = __expf(sc[kk][r] - mnew);
        sc[kk][r] = p;
        rs += p;
      }
      rs = half16_sum(rs);
      lst[r] = lst[r] * alpha + rs;
      mst[r] = mnew;
#pragma unroll
      for (int dt = 0; dt < 4; ++dt) o[dt][r] *= alpha;
    }

    // P (C-layout) -> wave-private LDS -> A-fragment layout (same-wave LDS
    // ops are in-order, no barrier needed).
    __bf16* Pw = &sP[wid * 16 * 64];
#pragma unroll
    for (int kk = 0; kk < 4; ++kk)
#pragma unroll
      for (int r = 0; r < 8; ++r)
        Pw[(r + sel * 8) * 64 + kk * 16 + nl] = (__bf16)sc[kk][r];

    v16bf pf0 = load_frag(Pw, 64, 0);
    v16bf pf1 = load_frag(Pw, 64, 32);

    // O += P @ V   (sVt is [d][key] so V fragments use the same pattern)
#pragma unroll
    for (int dt = 0; dt < 4; ++dt) {
      v16bf vf0 = load_frag(&sVt[dt * 16 * 64], 64, 0);
      v16bf vf1 = load_frag(&sVt[dt * 16 * 64], 64, 32);
      o[dt] = wmma_bf16(pf0, vf0, o[dt]);
      o[dt] = wmma_bf16(pf1, vf1, o[dt]);
    }
  }

  // normalize and write merged-head output (bf16, row = b*S+s, col = h*64+d)
  const int b = bh >> 4, h = bh & 15;
#pragma unroll
  for (int dt = 0; dt < 4; ++dt) {
#pragma unroll
    for (int r = 0; r < 8; ++r) {
      const int rowq = q0 + r + sel * 8;
      const float val = o[dt][r] / lst[r];
      out[(size_t)(b * SEQ + rowq) * D_MODEL + h * DK + dt * 16 + nl] =
          (__bf16)val;
    }
  }
}

// ---------------------------------------------------------------------------
// launch
// ---------------------------------------------------------------------------
extern "C" void kernel_launch(void* const* d_in, const int* in_sizes, int n_in,
                              void* d_out, int out_size, void* d_ws,
                              size_t ws_size, hipStream_t stream) {
  (void)in_sizes; (void)n_in; (void)out_size; (void)ws_size;
  const float* q  = (const float*)d_in[0];
  const float* k  = (const float*)d_in[1];
  const float* v  = (const float*)d_in[2];
  // d_in[3] is the causal tril mask; applied analytically in the kernel.
  const float* wq = (const float*)d_in[4];
  const float* bq = (const float*)d_in[5];
  const float* wk = (const float*)d_in[6];
  const float* bk = (const float*)d_in[7];
  const float* wv = (const float*)d_in[8];
  const float* bv = (const float*)d_in[9];
  const float* wo = (const float*)d_in[10];
  const float* bo = (const float*)d_in[11];

  const size_t N = (size_t)MTOT * D_MODEL;   // 4M elements
  __bf16* qh = (__bf16*)d_ws;                // [B,H,S,DK] bf16
  __bf16* kh = qh + N;
  __bf16* vh = kh + N;
  __bf16* ao = vh + N;                       // merged-head attn out, bf16

  dim3 gg(MTOT / 128, D_MODEL / 128);        // 32 x 8
  mha_gemm_wmma<float, 0><<<gg, 256, 0, stream>>>(q, wq, bq, qh);
  mha_gemm_wmma<float, 0><<<gg, 256, 0, stream>>>(k, wk, bk, kh);
  mha_gemm_wmma<float, 0><<<gg, 256, 0, stream>>>(v, wv, bv, vh);

  mha_flash_attn_wmma<<<dim3(SEQ / 128, BATCH * NH), 256, 0, stream>>>(qh, kh,
                                                                       vh, ao);

  mha_gemm_wmma<__bf16, 1><<<gg, 256, 0, stream>>>(ao, wo, bo, (float*)d_out);
}
